// Backbone_6133213299013
// MI455X (gfx1250) — matrix operand order repaired and checked by reading, not compiled, not run on hardware
//
#include <hip/hip_runtime.h>
#include <hip/hip_bf16.h>

#define N_AG   64
#define HOR    50
#define KM     6
#define DM     128
#define DURP1  11
#define HM1    49
#define PI_F   3.14159265358979323846f

typedef __attribute__((ext_vector_type(16))) _Float16 v16h;
typedef __attribute__((ext_vector_type(8)))  _Float16 v8h;
typedef __attribute__((ext_vector_type(8)))  float    v8f;

__device__ __forceinline__ float wrap_pi(float a) {
    // python-style mod: result in [0, 2pi)
    float r = fmodf(a + PI_F, 2.0f * PI_F);
    if (r < 0.0f) r += 2.0f * PI_F;
    return r - PI_F;
}

// FAM: 0 = a_embs, 1 = t2m_embs, 2 = F_embs, 3 = H_embs
template <int FAM>
__global__ __launch_bounds__(256)
void mlp_wmma_kernel(const float* __restrict__ W1, const float* __restrict__ b1,
                     const float* __restrict__ gamma, const float* __restrict__ beta,
                     const float* __restrict__ W2, const float* __restrict__ b2,
                     const float* __restrict__ vel_len, const float* __restrict__ vel_th,
                     const float* __restrict__ length, const float* __restrict__ width,
                     const float* __restrict__ position, const float* __restrict__ heading,
                     const int* __restrict__ a_batch, const int* __restrict__ agent_type,
                     const int* __restrict__ vis,
                     float* __restrict__ out, int numRows) {
    constexpr int HB_STRIDE  = 136;  // halves; 272B rows, 16B aligned, bank-skewed
    constexpr int WT_STRIDE  = 136;
    constexpr int OST_STRIDE = 132;  // floats; 528B rows, 16B aligned, bank-skewed
    constexpr int DIN = (FAM == 0) ? 5 : (FAM == 1) ? 4 : 3;

    // One arena; after the WMMA phase wt+hbuf are dead and the f32 output
    // staging buffer aliases them (33792 B <= 52224 B).
    __shared__ __align__(16) char smem[(128 * WT_STRIDE + 64 * HB_STRIDE) * 2];
    _Float16* wt   = (_Float16*)smem;                              // W2^T [n][k], f16
    _Float16* hbuf = (_Float16*)(smem + 128 * WT_STRIDE * 2);      // hidden, f16
    float*    ost  = (float*)smem;                                 // 64 x 132 f32 staging

    __shared__ float W1s[5 * 128];
    __shared__ float b1s[128], gs[128], bs[128], b2s[128];
    __shared__ float psum[64][4], psq[64][4];
    __shared__ float rowmask[64];

    const int tid = threadIdx.x;

    // ---- Phase 1: stage parameters into LDS (W2 transposed + f16) ----
    for (int i = tid; i < DIN * 128; i += 256) W1s[i] = W1[i];
    if (tid < 128) { b1s[tid] = b1[tid]; gs[tid] = gamma[tid]; bs[tid] = beta[tid]; b2s[tid] = b2[tid]; }
    for (int i = tid; i < 128 * 128; i += 256) {
        int k = i >> 7, n = i & 127;
        wt[n * WT_STRIDE + k] = (_Float16)W2[i];
    }
    __syncthreads();

    // ---- Phase 2: features + layer1 + LayerNorm + ReLU -> hbuf (f16) ----
    const int rowsBase = blockIdx.x * 64;
    const int lr = tid >> 2;          // local row 0..63
    const int q  = tid & 3;           // quarter of the 128 hidden units
    const int gr = rowsBase + lr;

    float f0 = 0.f, f1 = 0.f, f2 = 0.f, f3 = 0.f, f4 = 0.f;
    float mask = 0.f;
    if (gr < numRows) {
        if (FAM == 0) {
            int n = gr / HOR;
            f0 = vel_len[gr];
            f1 = vel_th[gr];
            f2 = length[n];
            f3 = width[n];
            f4 = (float)agent_type[n];
            mask = 1.0f;
        } else if (FAM == 1) {
            int tmp = gr;
            int dd = tmp % DURP1; tmp /= DURP1;
            int t = tmp % HOR;
            int n = tmp / HOR;
            int st  = t - dd;
            int stc = st < 0 ? 0 : st;
            float px = position[(n * HOR + stc) * 2 + 0], py = position[(n * HOR + stc) * 2 + 1];
            float cx = position[(n * HOR + t)   * 2 + 0], cy = position[(n * HOR + t)   * 2 + 1];
            float hs = heading[n * HOR + stc], hm = heading[n * HOR + t];
            float vx = px - cx, vy = py - cy;
            float c_ = cosf(hm), s_ = sinf(hm);
            float lx =  c_ * vx + s_ * vy;
            float ly = -s_ * vx + c_ * vy;
            f0 = sqrtf(lx * lx + ly * ly);
            f1 = atan2f(ly, lx);
            f2 = wrap_pi(hs - hm);
            f3 = -(float)dd;
            mask = (vis[n * HOR + stc] && vis[n * HOR + t] && (st >= 0)) ? 1.0f : 0.0f;
        } else {
            int tmp = gr;
            int j  = tmp % N_AG; tmp /= N_AG;
            int tt = tmp % HM1;
            int i  = tmp / HM1;
            float pix = position[(i * HOR + tt + 1) * 2 + 0], piy = position[(i * HOR + tt + 1) * 2 + 1];
            float pjx = position[(j * HOR + tt)     * 2 + 0], pjy = position[(j * HOR + tt)     * 2 + 1];
            float hi = heading[i * HOR + tt + 1], hj = heading[j * HOR + tt];
            float dx = pjx - pix, dy = pjy - piy;
            bool em  = (a_batch[i] == a_batch[j]) && (i != j) &&
                       (vis[i * HOR + tt + 1] != 0) && (vis[j * HOR + tt] != 0);
            bool dok = sqrtf(dx * dx + dy * dy) < 50.0f;
            mask = (em && dok) ? 1.0f : 0.0f;
            if (FAM == 2) {
                float vx = pix - pjx, vy = piy - pjy;           // _local(pos_i, pos_j, head_j)
                float c_ = cosf(hj), s_ = sinf(hj);
                float lx =  c_ * vx + s_ * vy;
                float ly = -s_ * vx + c_ * vy;
                f0 = sqrtf(lx * lx + ly * ly);
                f1 = atan2f(ly, lx);
                f2 = wrap_pi(hi - hj);
            } else {
                float vx = pjx - pix, vy = pjy - piy;           // _local(pos_j, pos_i, head_i)
                float c_ = cosf(hi), s_ = sinf(hi);
                float lx =  c_ * vx + s_ * vy;
                float ly = -s_ * vx + c_ * vy;
                f0 = sqrtf(lx * lx + ly * ly);
                f1 = atan2f(ly, lx);
                f2 = wrap_pi(hj - hi);
            }
        }
    }
    if (q == 0) rowmask[lr] = mask;

    float acc[32];
    float ps = 0.f, pq2 = 0.f;
#pragma unroll
    for (int u = 0; u < 32; ++u) {
        int col = q * 32 + u;
        float s = b1s[col];
        s += f0 * W1s[0 * 128 + col];
        s += f1 * W1s[1 * 128 + col];
        s += f2 * W1s[2 * 128 + col];
        if (DIN > 3) s += f3 * W1s[3 * 128 + col];
        if (DIN > 4) s += f4 * W1s[4 * 128 + col];
        acc[u] = s;
        ps  += s;
        pq2 += s * s;
    }
    psum[lr][q] = ps;
    psq[lr][q]  = pq2;
    __syncthreads();

    float mean = (psum[lr][0] + psum[lr][1] + psum[lr][2] + psum[lr][3]) * (1.0f / 128.0f);
    float var  = (psq[lr][0]  + psq[lr][1]  + psq[lr][2]  + psq[lr][3])  * (1.0f / 128.0f) - mean * mean;
    float inv  = rsqrtf(var + 1e-5f);
#pragma unroll
    for (int u = 0; u < 32; ++u) {
        int col = q * 32 + u;
        float hv = (acc[u] - mean) * inv * gs[col] + bs[col];
        hv = hv > 0.f ? hv : 0.f;                               // ReLU
        hbuf[lr * HB_STRIDE + col] = (_Float16)hv;
    }
    __syncthreads();

    // ---- Phase 3: second layer, 128x128 via v_wmma_f32_16x16x32_f16 ----
    const int wave = tid >> 5;
    const int lane = tid & 31;
    const int rt   = wave >> 1;          // row-tile 0..3
    const int cg   = wave & 1;           // column-tile group 0..1
    const int l15  = lane & 15;
    const int aoff = (lane < 16) ? 0 : 8;  // A-frag K sub-offset per lane half
    const int arow = rt * 16 + l15;

    v16h afrag[4];
#pragma unroll
    for (int kc = 0; kc < 4; ++kc) {
        int kb = kc * 32;
        const v8h lo = *(const v8h*)&hbuf[arow * HB_STRIDE + kb + aoff];
        const v8h hi = *(const v8h*)&hbuf[arow * HB_STRIDE + kb + aoff + 16];
        v16h a;
#pragma unroll
        for (int e = 0; e < 8; ++e) { a[e] = lo[e]; a[e + 8] = hi[e]; }
        afrag[kc] = a;
    }

    const int koffB = (lane < 16) ? 0 : 8;   // B-frag K sub-offset per lane half
    const int mbase = (lane < 16) ? 0 : 8;

    v8f accs[4];
#pragma unroll
    for (int c = 0; c < 4; ++c) {
        int ct   = cg * 4 + c;
        int ncol = ct * 16 + l15;
        float bias = b2s[ncol];
        v8f acc8;
#pragma unroll
        for (int e = 0; e < 8; ++e) acc8[e] = bias;
#pragma unroll
        for (int kc = 0; kc < 4; ++kc) {
            int kb = kc * 32 + koffB;
            const v8h blo = *(const v8h*)&wt[ncol * WT_STRIDE + kb];
            const v8h bhi = *(const v8h*)&wt[ncol * WT_STRIDE + kb + 16];
            v16h bfr;
#pragma unroll
            for (int e = 0; e < 8; ++e) { bfr[e] = blo[e]; bfr[e + 8] = bhi[e]; }
            acc8 = __builtin_amdgcn_wmma_f32_16x16x32_f16(
                false, afrag[kc], false, bfr, (short)0, acc8, false, false);
        }
        accs[c] = acc8;
    }

    // ---- Phase 4: transpose accumulators through LDS, then b128 streaming stores ----
    __syncthreads();   // all waves done reading wt/hbuf before aliasing writes
#pragma unroll
    for (int c = 0; c < 4; ++c) {
        int ncol = (cg * 4 + c) * 16 + l15;
#pragma unroll
        for (int e = 0; e < 8; ++e) {
            int lrow = rt * 16 + mbase + e;
            ost[lrow * OST_STRIDE + ncol] = accs[c][e] * rowmask[lrow];
        }
    }
    __syncthreads();

    // 64 rows x 32 float4 = 2048 float4 per block; 8 per thread, unconditional b128.
#pragma unroll
    for (int i = 0; i < 8; ++i) {
        int idx4 = tid + i * 256;
        int row  = idx4 >> 5;
        int c4   = idx4 & 31;
        int grow = rowsBase + row;
        if (grow < numRows) {
            float4 v = *(const float4*)&ost[row * OST_STRIDE + c4 * 4];
            *(float4*)&out[grow * DM + c4 * 4] = v;
        }
    }
}

// m_embs: broadcast mode_tokens (K,D) -> (H, N, K, D). Pure streaming store.
__global__ __launch_bounds__(256)
void mode_bcast_kernel(const float* __restrict__ tok, float* __restrict__ out) {
    int idx = blockIdx.x * blockDim.x + threadIdx.x;   // float4 index
    const int total4 = (HOR * N_AG * KM * DM) / 4;     // 614400
    if (idx < total4) {
        int e = (idx * 4) % (KM * DM);                 // inner (k,d) block repeats
        float4 v = *(const float4*)&tok[e];
        ((float4*)out)[idx] = v;
    }
}

extern "C" void kernel_launch(void* const* d_in, const int* in_sizes, int n_in,
                              void* d_out, int out_size, void* d_ws, size_t ws_size,
                              hipStream_t stream) {
    (void)in_sizes; (void)n_in; (void)out_size; (void)d_ws; (void)ws_size;

    const float* vel_len    = (const float*)d_in[0];
    const float* vel_th     = (const float*)d_in[1];
    const float* length     = (const float*)d_in[2];
    const float* width      = (const float*)d_in[3];
    const float* position   = (const float*)d_in[4];
    const float* heading    = (const float*)d_in[5];
    const int*   a_batch    = (const int*)d_in[6];
    const int*   agent_type = (const int*)d_in[7];
    const int*   vis        = (const int*)d_in[8];

    // params flatten in dict order: a, t2m, F, Hh (each W1,b1,g,b,W2,b2), then mode_tokens
    const float* aW1 = (const float*)d_in[9];
    const float* ab1 = (const float*)d_in[10];
    const float* ag  = (const float*)d_in[11];
    const float* ab  = (const float*)d_in[12];
    const float* aW2 = (const float*)d_in[13];
    const float* ab2 = (const float*)d_in[14];
    const float* tW1 = (const float*)d_in[15];
    const float* tb1 = (const float*)d_in[16];
    const float* tg  = (const float*)d_in[17];
    const float* tb  = (const float*)d_in[18];
    const float* tW2 = (const float*)d_in[19];
    const float* tb2 = (const float*)d_in[20];
    const float* fW1 = (const float*)d_in[21];
    const float* fb1 = (const float*)d_in[22];
    const float* fg  = (const float*)d_in[23];
    const float* fb  = (const float*)d_in[24];
    const float* fW2 = (const float*)d_in[25];
    const float* fb2 = (const float*)d_in[26];
    const float* hW1 = (const float*)d_in[27];
    const float* hb1 = (const float*)d_in[28];
    const float* hg  = (const float*)d_in[29];
    const float* hb  = (const float*)d_in[30];
    const float* hW2 = (const float*)d_in[31];
    const float* hb2 = (const float*)d_in[32];
    const float* tok = (const float*)d_in[33];

    float* out = (float*)d_out;
    // Concatenated outputs (flat float counts):
    float* out_a   = out;                // N*H*D          = 409600
    float* out_m   = out + 409600;       // H*N*K*D        = 2457600
    float* out_t2m = out + 2867200;      // N*H*11*D       = 4505600
    float* out_F   = out + 7372800;      // N*49*N*D       = 25690112
    float* out_H   = out + 33062912;     // N*49*N*D       = 25690112

    const int rows_a   = N_AG * HOR;               // 3200
    const int rows_t2m = N_AG * HOR * DURP1;       // 35200
    const int rows_fh  = N_AG * HM1 * N_AG;        // 200704

    dim3 blk(256);

    mode_bcast_kernel<<<dim3((HOR * N_AG * KM * DM / 4 + 255) / 256), blk, 0, stream>>>(tok, out_m);

    mlp_wmma_kernel<0><<<dim3((rows_a + 63) / 64), blk, 0, stream>>>(
        aW1, ab1, ag, ab, aW2, ab2,
        vel_len, vel_th, length, width, position, heading,
        a_batch, agent_type, vis, out_a, rows_a);

    mlp_wmma_kernel<1><<<dim3((rows_t2m + 63) / 64), blk, 0, stream>>>(
        tW1, tb1, tg, tb, tW2, tb2,
        vel_len, vel_th, length, width, position, heading,
        a_batch, agent_type, vis, out_t2m, rows_t2m);

    mlp_wmma_kernel<2><<<dim3((rows_fh + 63) / 64), blk, 0, stream>>>(
        fW1, fb1, fg, fb, fW2, fb2,
        vel_len, vel_th, length, width, position, heading,
        a_batch, agent_type, vis, out_F, rows_fh);

    mlp_wmma_kernel<3><<<dim3((rows_fh + 63) / 64), blk, 0, stream>>>(
        hW1, hb1, hg, hb, hW2, hb2,
        vel_len, vel_th, length, width, position, heading,
        a_batch, agent_type, vis, out_H, rows_fh);
}
